// MultiHeadAttention_14087492731518
// MI455X (gfx1250) — matrix operand order repaired and checked by reading, not compiled, not run on hardware
//
#include <hip/hip_runtime.h>
#include <hip/hip_bf16.h>

#define D_MODEL 1024
#define N_HEADS 16
#define D_K     64
#define SEQ     2048
#define BATCH   2
#define M_TOTAL (BATCH * SEQ)   // 4096
#define LN_THETA 9.2103403719761836f   // ln(10000)

typedef __attribute__((ext_vector_type(16))) __bf16 v16bf;
typedef __attribute__((ext_vector_type(8)))  __bf16 v8bf;
typedef __attribute__((ext_vector_type(4)))  __bf16 v4bf;
typedef __attribute__((ext_vector_type(8)))  float  v8f;
typedef __attribute__((ext_vector_type(4)))  int    v4i;

#ifndef TRY_ASYNC
#define TRY_ASYNC 1
#endif
#if TRY_ASYNC && defined(__has_builtin)
# if __has_builtin(__builtin_amdgcn_global_load_async_to_lds_b128) && \
     __has_builtin(__builtin_amdgcn_s_wait_asynccnt)
#  define HAVE_ASYNC_LDS 1
# else
#  define HAVE_ASYNC_LDS 0
# endif
#else
# define HAVE_ASYNC_LDS 0
#endif

static __device__ inline __bf16 f2bf(float f) {
    union { float f; unsigned u; } v; v.f = f;
    unsigned r = v.u + 0x7FFFu + ((v.u >> 16) & 1u);   // round-to-nearest-even
    unsigned short h = (unsigned short)(r >> 16);
    __bf16 b;
    __builtin_memcpy(&b, &h, sizeof(h));
    return b;
}

static __device__ inline v16bf cmb(v8bf lo, v8bf hi) {
    return __builtin_shufflevector(lo, hi, 0,1,2,3,4,5,6,7,8,9,10,11,12,13,14,15);
}

static __device__ inline float rmax16(float v) {
    #pragma unroll
    for (int m = 1; m < 16; m <<= 1) v = fmaxf(v, __shfl_xor(v, m, 32));
    return v;
}
static __device__ inline float rsum16(float v) {
    #pragma unroll
    for (int m = 1; m < 16; m <<= 1) v += __shfl_xor(v, m, 32);
    return v;
}

// ---------------------------------------------------------------------------
// Kernel 0: one-shot fp32 -> bf16 (RNE).
// ---------------------------------------------------------------------------
__global__ __launch_bounds__(256)
void f32_to_bf16_kernel(const float* __restrict__ in, __bf16* __restrict__ out, int n4)
{
    const int i = blockIdx.x * 256 + threadIdx.x;
    if (i < n4) {
        float4 v = ((const float4*)in)[i];
        v4bf o;
        o[0] = f2bf(v.x); o[1] = f2bf(v.y); o[2] = f2bf(v.z); o[3] = f2bf(v.w);
        ((v4bf*)out)[i] = o;
    }
}

// ---------------------------------------------------------------------------
// Kernel 1: Y = X @ W^T (bf16 in, fp32 accum), software-pipelined K loop,
// fused RoPE epilogue. Wave tile = 16M x 64N.
// ---------------------------------------------------------------------------
__global__ __launch_bounds__(256)
void proj_rope_kernel(const __bf16* __restrict__ X, const __bf16* __restrict__ W,
                      const int* __restrict__ pos, __bf16* __restrict__ out, int mode)
{
    const int lane = threadIdx.x & 31;
    const int wave = threadIdx.x >> 5;
    const int l16  = lane & 15;
    const int hi   = lane >> 4;

    const int m  = blockIdx.x * 16 + l16;                // row in [0, M_TOTAL)
    const int n0 = blockIdx.y * 512 + wave * 64;         // wave col base
    const __bf16* xrow = X + (size_t)m * D_MODEL + hi * 16;
    const __bf16* wrow0 = W + (size_t)(n0 +  0 + l16) * D_MODEL + hi * 16;
    const __bf16* wrow1 = W + (size_t)(n0 + 16 + l16) * D_MODEL + hi * 16;
    const __bf16* wrow2 = W + (size_t)(n0 + 32 + l16) * D_MODEL + hi * 16;
    const __bf16* wrow3 = W + (size_t)(n0 + 48 + l16) * D_MODEL + hi * 16;

    // prologue: fragments for kb = 0
    v16bf af = cmb(*(const v8bf*)(xrow), *(const v8bf*)(xrow + 8));
    v16bf b0 = cmb(*(const v8bf*)(wrow0), *(const v8bf*)(wrow0 + 8));
    v16bf b1 = cmb(*(const v8bf*)(wrow1), *(const v8bf*)(wrow1 + 8));
    v16bf b2 = cmb(*(const v8bf*)(wrow2), *(const v8bf*)(wrow2 + 8));
    v16bf b3 = cmb(*(const v8bf*)(wrow3), *(const v8bf*)(wrow3 + 8));

    v8f acc0 = {}, acc1 = {}, acc2 = {}, acc3 = {};
    #pragma unroll 2
    for (int kb = 0; kb < D_MODEL; kb += 32) {
        // issue next iteration's loads first (wrap keeps it branchless)
        const int kn = (kb + 32) & (D_MODEL - 1);
        v16bf afn = cmb(*(const v8bf*)(xrow + kn), *(const v8bf*)(xrow + kn + 8));
        v16bf b0n = cmb(*(const v8bf*)(wrow0 + kn), *(const v8bf*)(wrow0 + kn + 8));
        v16bf b1n = cmb(*(const v8bf*)(wrow1 + kn), *(const v8bf*)(wrow1 + kn + 8));
        v16bf b2n = cmb(*(const v8bf*)(wrow2 + kn), *(const v8bf*)(wrow2 + kn + 8));
        v16bf b3n = cmb(*(const v8bf*)(wrow3 + kn), *(const v8bf*)(wrow3 + kn + 8));
        __builtin_prefetch(xrow + kb + 256, 0, 1);
        // consume current fragments (loads issued one iteration ago)
        acc0 = __builtin_amdgcn_wmma_f32_16x16x32_bf16(false, af, false, b0, (short)0, acc0, false, false);
        acc1 = __builtin_amdgcn_wmma_f32_16x16x32_bf16(false, af, false, b1, (short)0, acc1, false, false);
        acc2 = __builtin_amdgcn_wmma_f32_16x16x32_bf16(false, af, false, b2, (short)0, acc2, false, false);
        acc3 = __builtin_amdgcn_wmma_f32_16x16x32_bf16(false, af, false, b3, (short)0, acc3, false, false);
        af = afn; b0 = b0n; b1 = b1n; b2 = b2n; b3 = b3n;
    }

    // Epilogue: C/D layout row = tile + r + hi*8, col = n0 + t*16 + l16
    const int head = n0 >> 6;                 // n0 multiple of 64
    #pragma unroll
    for (int t = 0; t < 4; ++t) {
        v8f acc = (t == 0) ? acc0 : (t == 1) ? acc1 : (t == 2) ? acc2 : acc3;
        const int n = n0 + t * 16 + l16;
        const int d = n & 63;
        #pragma unroll
        for (int r = 0; r < 8; ++r) {
            const int row = blockIdx.x * 16 + r + hi * 8;
            const int b   = row >> 11;          // / SEQ
            const int s   = row & (SEQ - 1);
            float v = acc[r];
            if (mode != 2) {                    // RoPE for q/k
                const int i2 = (d >> 1) << 1;   // 2*(d/2)
                const float invf = __expf(-(float)i2 * (LN_THETA / 64.0f));
                const float ang  = (float)pos[s] * invf;
                float sn, cs; __sincosf(ang, &sn, &cs);
                const float other = __shfl_xor(v, 1, 32);
                v = ((n & 1) == 0) ? (v * cs - other * sn)    // even: xe*cos - xo*sin
                                   : (other * sn + v * cs);   // odd : xe*sin + xo*cos
            }
            size_t idx = (mode == 2)
                ? ((((size_t)b * N_HEADS + head) * D_K + d) * SEQ + s)   // vT [B,H,d,S]
                : ((((size_t)b * N_HEADS + head) * SEQ + s) * D_K + d);  // q/k [B,H,S,d]
            out[idx] = f2bf(v);
        }
    }
}

// ---------------------------------------------------------------------------
// Kernel 2: causal flash attention, K-fragment stream double-buffered.
// ---------------------------------------------------------------------------
__global__ __launch_bounds__(256)
void flash_attn_kernel(const __bf16* __restrict__ qh, const __bf16* __restrict__ kh,
                       const __bf16* __restrict__ vT, __bf16* __restrict__ attn)
{
    __shared__ __bf16 lds[8][16 * 32];      // per-wave 16x32 P tile (8 KB)
#if HAVE_ASYNC_LDS
    __shared__ __bf16 qtile[8][16 * D_K];   // per-wave async-staged Q tile (16 KB)
#endif

    const int lane = threadIdx.x & 31;
    const int wave = threadIdx.x >> 5;
    const int l16  = lane & 15;
    const int hi   = lane >> 4;
    const int b    = blockIdx.z;
    const int h    = blockIdx.y;
    const int qb   = blockIdx.x * 128 + wave * 16;

    const __bf16* qbh = qh + (((size_t)b * N_HEADS + h) * SEQ) * D_K;
    const __bf16* kbh = kh + (((size_t)b * N_HEADS + h) * SEQ) * D_K;
    const __bf16* vbh = vT + (((size_t)b * N_HEADS + h) * D_K) * SEQ;

    // ---- Q fragments for K-steps d=[0,32) and d=[32,64) ----
    v16bf qf0, qf1;
#if HAVE_ASYNC_LDS
    {
        // Q tile rows qb..qb+15 are 2 KB contiguous in [B,H,S,D_K]; stream them
        // to LDS with the gfx1250 async-copy path (tracked by ASYNCcnt).
        typedef __attribute__((address_space(1))) v4i gv4i;
        typedef __attribute__((address_space(3))) v4i lv4i;
        gv4i* g = (gv4i*)(unsigned long long)(const void*)(qbh + (size_t)qb * D_K);
        lv4i* l = (lv4i*)(unsigned)(size_t)(const void*)(&qtile[wave][0]);
        #pragma unroll
        for (int t = 0; t < 4; ++t)
            __builtin_amdgcn_global_load_async_to_lds_b128(
                g + lane + t * 32, l + lane + t * 32, 0, 0);
        __builtin_amdgcn_s_wait_asynccnt(0);
        const __bf16* qrow = &qtile[wave][l16 * D_K];
        qf0 = cmb(*(const v8bf*)(qrow +  0 + hi * 16), *(const v8bf*)(qrow +  8 + hi * 16));
        qf1 = cmb(*(const v8bf*)(qrow + 32 + hi * 16), *(const v8bf*)(qrow + 40 + hi * 16));
    }
#else
    {
        const __bf16* qrow = qbh + (size_t)(qb + l16) * D_K;
        qf0 = cmb(*(const v8bf*)(qrow +  0 + hi * 16), *(const v8bf*)(qrow +  8 + hi * 16));
        qf1 = cmb(*(const v8bf*)(qrow + 32 + hi * 16), *(const v8bf*)(qrow + 40 + hi * 16));
    }
#endif

    v8f o0 = {}, o1 = {}, o2 = {}, o3 = {};
    float mrow[8], lrow[8];
    #pragma unroll
    for (int r = 0; r < 8; ++r) { mrow[r] = -1e30f; lrow[r] = 0.0f; }

    // prologue: K fragments for jb = 0
    const __bf16* k0 = kbh + (size_t)(l16) * D_K + hi * 16;
    const __bf16* k1 = kbh + (size_t)(16 + l16) * D_K + hi * 16;
    v16bf kf00 = cmb(*(const v8bf*)(k0 +  0), *(const v8bf*)(k0 +  8));
    v16bf kf01 = cmb(*(const v8bf*)(k0 + 32), *(const v8bf*)(k0 + 40));
    v16bf kf10 = cmb(*(const v8bf*)(k1 +  0), *(const v8bf*)(k1 +  8));
    v16bf kf11 = cmb(*(const v8bf*)(k1 + 32), *(const v8bf*)(k1 + 40));

    for (int jb = 0; jb <= qb; jb += 32) {
        v8f c0 = {}, c1 = {};
        c0 = __builtin_amdgcn_wmma_f32_16x16x32_bf16(false, qf0, false, kf00, (short)0, c0, false, false);
        c0 = __builtin_amdgcn_wmma_f32_16x16x32_bf16(false, qf1, false, kf01, (short)0, c0, false, false);
        c1 = __builtin_amdgcn_wmma_f32_16x16x32_bf16(false, qf0, false, kf10, (short)0, c1, false, false);
        c1 = __builtin_amdgcn_wmma_f32_16x16x32_bf16(false, qf1, false, kf11, (short)0, c1, false, false);

        // issue next key-block's K loads now; they overlap softmax + PV below
        {
            const int jn = (jb + 32 <= qb) ? (jb + 32) : jb;
            const __bf16* n0p = kbh + (size_t)(jn      + l16) * D_K + hi * 16;
            const __bf16* n1p = kbh + (size_t)(jn + 16 + l16) * D_K + hi * 16;
            v16bf t00 = cmb(*(const v8bf*)(n0p +  0), *(const v8bf*)(n0p +  8));
            v16bf t01 = cmb(*(const v8bf*)(n0p + 32), *(const v8bf*)(n0p + 40));
            v16bf t10 = cmb(*(const v8bf*)(n1p +  0), *(const v8bf*)(n1p +  8));
            v16bf t11 = cmb(*(const v8bf*)(n1p + 32), *(const v8bf*)(n1p + 40));

            // causal mask + online softmax over this 32-key block
            const int col0 = jb + l16, col1 = jb + 16 + l16;
            #pragma unroll
            for (int r = 0; r < 8; ++r) {
                const int row = qb + r + hi * 8;
                float s0 = c0[r] * 0.125f; if (col0 > row) s0 = -1e30f;
                float s1 = c1[r] * 0.125f; if (col1 > row) s1 = -1e30f;
                const float mx = rmax16(fmaxf(s0, s1));
                const float mn = fmaxf(mrow[r], mx);
                const float alpha = __expf(mrow[r] - mn);
                const float e0 = __expf(s0 - mn);
                const float e1 = __expf(s1 - mn);
                const float rs = rsum16(e0 + e1);
                lrow[r] = lrow[r] * alpha + rs;
                mrow[r] = mn;
                o0[r] *= alpha; o1[r] *= alpha; o2[r] *= alpha; o3[r] *= alpha;
                lds[wave][(r + hi * 8) * 32 +      l16] = f2bf(e0);
                lds[wave][(r + hi * 8) * 32 + 16 + l16] = f2bf(e1);
            }

            // P A-fragment: row = l16, K = i+hi*16 over 32 keys
            const __bf16* pr = &lds[wave][l16 * 32 + hi * 16];
            v16bf pf = cmb(*(const v8bf*)(pr), *(const v8bf*)(pr + 8));

            // V B-fragments from vT[B,H,d,S]: lane col = d, contiguous keys
            const __bf16* v0p = vbh + (size_t)( 0 + l16) * SEQ + jb + hi * 16;
            const __bf16* v1p = vbh + (size_t)(16 + l16) * SEQ + jb + hi * 16;
            const __bf16* v2p = vbh + (size_t)(32 + l16) * SEQ + jb + hi * 16;
            const __bf16* v3p = vbh + (size_t)(48 + l16) * SEQ + jb + hi * 16;
            v16bf vf0 = cmb(*(const v8bf*)(v0p), *(const v8bf*)(v0p + 8));
            v16bf vf1 = cmb(*(const v8bf*)(v1p), *(const v8bf*)(v1p + 8));
            v16bf vf2 = cmb(*(const v8bf*)(v2p), *(const v8bf*)(v2p + 8));
            v16bf vf3 = cmb(*(const v8bf*)(v3p), *(const v8bf*)(v3p + 8));

            o0 = __builtin_amdgcn_wmma_f32_16x16x32_bf16(false, pf, false, vf0, (short)0, o0, false, false);
            o1 = __builtin_amdgcn_wmma_f32_16x16x32_bf16(false, pf, false, vf1, (short)0, o1, false, false);
            o2 = __builtin_amdgcn_wmma_f32_16x16x32_bf16(false, pf, false, vf2, (short)0, o2, false, false);
            o3 = __builtin_amdgcn_wmma_f32_16x16x32_bf16(false, pf, false, vf3, (short)0, o3, false, false);

            kf00 = t00; kf01 = t01; kf10 = t10; kf11 = t11;
        }
    }

    #pragma unroll
    for (int r = 0; r < 8; ++r) {
        const int s = qb + r + hi * 8;
        const float inv_l = 1.0f / lrow[r];
        __bf16* orow = attn + ((size_t)b * SEQ + s) * D_MODEL + h * D_K + l16;
        orow[ 0] = f2bf(o0[r] * inv_l);
        orow[16] = f2bf(o1[r] * inv_l);
        orow[32] = f2bf(o2[r] * inv_l);
        orow[48] = f2bf(o3[r] * inv_l);
    }
}

// ---------------------------------------------------------------------------
// Kernel 3: out = attn @ w_o^T (bf16 in, fp32 out), software-pipelined.
// ---------------------------------------------------------------------------
__global__ __launch_bounds__(256)
void out_proj_kernel(const __bf16* __restrict__ A, const __bf16* __restrict__ W,
                     float* __restrict__ out)
{
    const int lane = threadIdx.x & 31;
    const int wave = threadIdx.x >> 5;
    const int l16  = lane & 15;
    const int hi   = lane >> 4;

    const int m  = blockIdx.x * 16 + l16;
    const int n0 = blockIdx.y * 512 + wave * 64;
    const __bf16* arow  = A + (size_t)m * D_MODEL + hi * 16;
    const __bf16* wrow0 = W + (size_t)(n0 +  0 + l16) * D_MODEL + hi * 16;
    const __bf16* wrow1 = W + (size_t)(n0 + 16 + l16) * D_MODEL + hi * 16;
    const __bf16* wrow2 = W + (size_t)(n0 + 32 + l16) * D_MODEL + hi * 16;
    const __bf16* wrow3 = W + (size_t)(n0 + 48 + l16) * D_MODEL + hi * 16;

    v16bf af = cmb(*(const v8bf*)(arow), *(const v8bf*)(arow + 8));
    v16bf b0 = cmb(*(const v8bf*)(wrow0), *(const v8bf*)(wrow0 + 8));
    v16bf b1 = cmb(*(const v8bf*)(wrow1), *(const v8bf*)(wrow1 + 8));
    v16bf b2 = cmb(*(const v8bf*)(wrow2), *(const v8bf*)(wrow2 + 8));
    v16bf b3 = cmb(*(const v8bf*)(wrow3), *(const v8bf*)(wrow3 + 8));

    v8f acc0 = {}, acc1 = {}, acc2 = {}, acc3 = {};
    #pragma unroll 2
    for (int kb = 0; kb < D_MODEL; kb += 32) {
        const int kn = (kb + 32) & (D_MODEL - 1);
        v16bf afn = cmb(*(const v8bf*)(arow + kn), *(const v8bf*)(arow + kn + 8));
        v16bf b0n = cmb(*(const v8bf*)(wrow0 + kn), *(const v8bf*)(wrow0 + kn + 8));
        v16bf b1n = cmb(*(const v8bf*)(wrow1 + kn), *(const v8bf*)(wrow1 + kn + 8));
        v16bf b2n = cmb(*(const v8bf*)(wrow2 + kn), *(const v8bf*)(wrow2 + kn + 8));
        v16bf b3n = cmb(*(const v8bf*)(wrow3 + kn), *(const v8bf*)(wrow3 + kn + 8));
        __builtin_prefetch(arow + kb + 256, 0, 1);
        acc0 = __builtin_amdgcn_wmma_f32_16x16x32_bf16(false, af, false, b0, (short)0, acc0, false, false);
        acc1 = __builtin_amdgcn_wmma_f32_16x16x32_bf16(false, af, false, b1, (short)0, acc1, false, false);
        acc2 = __builtin_amdgcn_wmma_f32_16x16x32_bf16(false, af, false, b2, (short)0, acc2, false, false);
        acc3 = __builtin_amdgcn_wmma_f32_16x16x32_bf16(false, af, false, b3, (short)0, acc3, false, false);
        af = afn; b0 = b0n; b1 = b1n; b2 = b2n; b3 = b3n;
    }

    #pragma unroll
    for (int t = 0; t < 4; ++t) {
        v8f acc = (t == 0) ? acc0 : (t == 1) ? acc1 : (t == 2) ? acc2 : acc3;
        const int n = n0 + t * 16 + l16;
        #pragma unroll
        for (int r = 0; r < 8; ++r) {
            const int row = blockIdx.x * 16 + r + hi * 8;
            out[(size_t)row * D_MODEL + n] = acc[r];
        }
    }
}

// ---------------------------------------------------------------------------
extern "C" void kernel_launch(void* const* d_in, const int* in_sizes, int n_in,
                              void* d_out, int out_size, void* d_ws, size_t ws_size,
                              hipStream_t stream)
{
    const float* Q  = (const float*)d_in[0];
    const float* K  = (const float*)d_in[1];
    const float* V  = (const float*)d_in[2];
    const float* wq = (const float*)d_in[3];
    const float* wk = (const float*)d_in[4];
    const float* wv = (const float*)d_in[5];
    const float* wo = (const float*)d_in[6];
    // d_in[7] = mask (causal, implemented analytically)
    const int* pos  = (const int*)d_in[8];

    const size_t inElems = (size_t)M_TOTAL * D_MODEL;               // 4 Mi
    const size_t wElems  = (size_t)D_MODEL * D_MODEL;               // 1 Mi
    const size_t hElems  = (size_t)BATCH * N_HEADS * SEQ * D_K;     // 4 Mi

    __bf16* Qb  = (__bf16*)d_ws;
    __bf16* Kb  = Qb  + inElems;
    __bf16* Vb  = Kb  + inElems;
    __bf16* wqb = Vb  + inElems;
    __bf16* wkb = wqb + wElems;
    __bf16* wvb = wkb + wElems;
    __bf16* wob = wvb + wElems;
    __bf16* qhw = wob + wElems;
    __bf16* khw = qhw + hElems;
    __bf16* vtw = khw + hElems;
    __bf16* att = vtw + hElems;                                     // [B*S][D_MODEL]

    const int in4 = (int)(inElems / 4), w4 = (int)(wElems / 4);
    f32_to_bf16_kernel<<<in4 / 256, 256, 0, stream>>>(Q,  Qb,  in4);
    f32_to_bf16_kernel<<<in4 / 256, 256, 0, stream>>>(K,  Kb,  in4);
    f32_to_bf16_kernel<<<in4 / 256, 256, 0, stream>>>(V,  Vb,  in4);
    f32_to_bf16_kernel<<<w4  / 256, 256, 0, stream>>>(wq, wqb, w4);
    f32_to_bf16_kernel<<<w4  / 256, 256, 0, stream>>>(wk, wkb, w4);
    f32_to_bf16_kernel<<<w4  / 256, 256, 0, stream>>>(wv, wvb, w4);
    f32_to_bf16_kernel<<<w4  / 256, 256, 0, stream>>>(wo, wob, w4);

    dim3 blk(256);
    dim3 pgrid(M_TOTAL / 16, D_MODEL / 512);
    proj_rope_kernel<<<pgrid, blk, 0, stream>>>(Qb, wqb, pos, qhw, 0);
    proj_rope_kernel<<<pgrid, blk, 0, stream>>>(Kb, wkb, pos, khw, 1);
    proj_rope_kernel<<<pgrid, blk, 0, stream>>>(Vb, wvb, pos, vtw, 2);

    dim3 agrid(SEQ / 128, N_HEADS, BATCH);
    flash_attn_kernel<<<agrid, blk, 0, stream>>>(qhw, khw, vtw, att);

    out_proj_kernel<<<pgrid, blk, 0, stream>>>(att, wob, (float*)d_out);
}
